// MultiHeadAttention_81088982548895
// MI455X (gfx1250) — matrix operand order repaired and checked by reading, not compiled, not run on hardware
//
#include <hip/hip_runtime.h>
#include <hip/hip_bf16.h>

// ---------------------------------------------------------------------------
// CDNA5 (gfx1250, wave32) sliding-window MHA via v_wmma_f32_16x16x32_f16.
// Pipeline: f32->f16 convert -> QKV GEMM (scatter Q,K, V^T) -> windowed
// attention (workgroup-shared K tile in LDS, QK^T WMMA, branchless mask, wave
// softmax, PV WMMA via LDS-staged P reusing the K buffer) -> out GEMM.
// ---------------------------------------------------------------------------

typedef __attribute__((ext_vector_type(8)))  _Float16 v8h;
typedef __attribute__((ext_vector_type(16))) _Float16 v16h;
typedef __attribute__((ext_vector_type(8)))  float    v8f;

#define SEQ    2048
#define EMB    512
#define HEADS  8
#define HD     64
#define NT     10            // 10 key tiles of 16 => 160 keys per wave
#define KROWS  272           // workgroup key rows: [qb-64, qb+207]
#define KPAD   72            // K row pitch (f16): 144B = 36 banks, conflict-free
#define PPAD   168           // P row pitch (f16): 336B = 84 banks, conflict-free
#define NEG_INF (-__builtin_huge_valf())

__device__ __forceinline__ v8h ldg8(const _Float16* base, unsigned off) {
  return *reinterpret_cast<const v8h*>(base + off);
}
__device__ __forceinline__ v8h lds8(const _Float16* p) {
  return *reinterpret_cast<const v8h*>(p);
}
__device__ __forceinline__ v16h cat16(v8h a, v8h b) {
  return __builtin_shufflevector(a, b, 0,1,2,3,4,5,6,7,8,9,10,11,12,13,14,15);
}
__device__ __forceinline__ v8f wmma16(v16h a, v16h b, v8f c) {
  // emits v_wmma_f32_16x16x32_f16 (probe-confirmed 8-arg signature)
  return __builtin_amdgcn_wmma_f32_16x16x32_f16(false, a, false, b,
                                                (short)0, c, false, false);
}

// ---------------------------------------------------------------------------
// Vectorized f32 -> f16 (8 elems/thread; n must be a multiple of 8 - true here)
__global__ void cvt_f32_f16(const float* __restrict__ src,
                            _Float16* __restrict__ dst, int n8) {
  int i = blockIdx.x * blockDim.x + threadIdx.x;
  int stride = gridDim.x * blockDim.x;
  for (; i < n8; i += stride) {
    const float4 a = reinterpret_cast<const float4*>(src)[2 * i];
    const float4 b = reinterpret_cast<const float4*>(src)[2 * i + 1];
    v8h o;
    o[0] = (_Float16)a.x; o[1] = (_Float16)a.y;
    o[2] = (_Float16)a.z; o[3] = (_Float16)a.w;
    o[4] = (_Float16)b.x; o[5] = (_Float16)b.y;
    o[6] = (_Float16)b.z; o[7] = (_Float16)b.w;
    reinterpret_cast<v8h*>(dst)[i] = o;
  }
}

// ---------------------------------------------------------------------------
// C[M,N] = A[M,K](row-major f16) x Bw[N,K](row-major f16, i.e. B^T) + bias.
// Workgroup tile 128x64, 8 waves of 32x32 (2x2 WMMA tiles), K-step 32.
// MODE 0: store f32 to Cout.  MODE 1: QKV scatter epilogue.
template <int MODE>
__global__ __launch_bounds__(256)
void gemm_wmma(const _Float16* __restrict__ A, int lda,
               const _Float16* __restrict__ Bw, int ldb,
               const float* __restrict__ bias, int K,
               float* __restrict__ Cout, int ldc,
               _Float16* __restrict__ Qo, _Float16* __restrict__ Ko,
               _Float16* __restrict__ Vt, int Sdim) {
  const int lane = threadIdx.x & 31;
  const int wid  = threadIdx.x >> 5;
  const int r    = lane & 15;
  const int half = lane >> 4;
  const int row0 = blockIdx.x * 128 + (wid & 3) * 32;
  const int col0 = blockIdx.y * 64  + (wid >> 2) * 32;

  const unsigned arow = (unsigned)(row0 + r) * lda + half * 16;
  const unsigned brow = (unsigned)(col0 + r) * ldb + half * 16;

  v8f acc[2][2] = {};
  for (int k0 = 0; k0 < K; k0 += 32) {
    const unsigned a0o = arow + k0;
    const unsigned a1o = a0o + 16u * lda;
    const unsigned b0o = brow + k0;
    const unsigned b1o = b0o + 16u * ldb;
    if (k0 + 32 < K) {  // cover next K-step (global_prefetch_b8)
      __builtin_prefetch(A + a0o + 32, 0, 0);
      __builtin_prefetch(Bw + b0o + 32, 0, 0);
    }
    v16h a0 = cat16(ldg8(A, a0o), ldg8(A, a0o + 8));
    v16h a1 = cat16(ldg8(A, a1o), ldg8(A, a1o + 8));
    v16h b0 = cat16(ldg8(Bw, b0o), ldg8(Bw, b0o + 8));
    v16h b1 = cat16(ldg8(Bw, b1o), ldg8(Bw, b1o + 8));
    acc[0][0] = wmma16(a0, b0, acc[0][0]);
    acc[0][1] = wmma16(a0, b1, acc[0][1]);
    acc[1][0] = wmma16(a1, b0, acc[1][0]);
    acc[1][1] = wmma16(a1, b1, acc[1][1]);
  }

  #pragma unroll
  for (int im = 0; im < 2; im++) {
    #pragma unroll
    for (int in_ = 0; in_ < 2; in_++) {
      const int mbase = row0 + im * 16 + half * 8;  // C row = vgpr + 8*half
      const int n = col0 + in_ * 16 + r;            // C col = lane&15
      const float bn = bias[n];
      if (MODE == 0) {
        #pragma unroll
        for (int rr = 0; rr < 8; rr++)
          Cout[(unsigned)(mbase + rr) * ldc + n] = acc[im][in_][rr] + bn;
      } else {
        const int h = n / 192;
        const int j = n - h * 192;
        const int which = j >> 6;
        const int d = j & 63;
        if (which == 0) {        // Q, pre-scaled by 1/sqrt(64)
          _Float16* dst = Qo + (unsigned)(h * Sdim + mbase) * HD + d;
          #pragma unroll
          for (int rr = 0; rr < 8; rr++)
            dst[rr * HD] = (_Float16)((acc[im][in_][rr] + bn) * 0.125f);
        } else if (which == 1) { // K
          _Float16* dst = Ko + (unsigned)(h * Sdim + mbase) * HD + d;
          #pragma unroll
          for (int rr = 0; rr < 8; rr++)
            dst[rr * HD] = (_Float16)(acc[im][in_][rr] + bn);
        } else {                 // V transposed: Vt[h][d][s]
          _Float16* dst = Vt + (unsigned)(h * HD + d) * Sdim + mbase;
          #pragma unroll
          for (int rr = 0; rr < 8; rr++)
            dst[rr] = (_Float16)(acc[im][in_][rr] + bn);
        }
      }
    }
  }
}

// ---------------------------------------------------------------------------
// Sliding-window attention. grid = (SEQ/128, HEADS), 256 threads (8 waves).
// Wave w owns queries [qw, qw+16); its 10 key tiles live at LDS rows
// [16*w, 16*w+160) of the workgroup tile [qb-64, qb+207] (272 rows).
__global__ __launch_bounds__(256)
void attn_wmma(const _Float16* __restrict__ Qh, const _Float16* __restrict__ Kh,
               const _Float16* __restrict__ Vt, const int* __restrict__ pm,
               _Float16* __restrict__ vals) {
  // One buffer, two lives: K tile (272 x KPAD f16 = 39168B) during scores,
  // then P tiles (8 x 16 x PPAD f16 = 43008B) after the barrier.
  __shared__ __align__(16) char smem[8 * 16 * PPAD * 2];
  _Float16* Ksh = reinterpret_cast<_Float16*>(smem);
  _Float16* Pb  = reinterpret_cast<_Float16*>(smem);

  const int h    = blockIdx.y;
  const int lane = threadIdx.x & 31;
  const int wid  = threadIdx.x >> 5;
  const int r    = lane & 15;
  const int half = lane >> 4;
  const int qb   = blockIdx.x * 128;
  const int qw   = qb + wid * 16;
  const int tstart = qw - 64;
  const int cdiag  = r - half * 8;  // lane-local part of the band predicate

  // Uniform per-head bases -> SGPRs; all indexing below is 32-bit.
  const _Float16* __restrict__ Qb = Qh + (size_t)h * (SEQ * HD);
  const _Float16* __restrict__ Kb = Kh + (size_t)h * (SEQ * HD);
  const _Float16* __restrict__ Vb = Vt + (size_t)h * (HD * SEQ);

  // ---- cooperative K tile load: rows [qb-64, qb+207], clamped ----
  for (int idx = threadIdx.x; idx < KROWS * 8; idx += 256) {
    const int row   = idx >> 3;       // 0..271
    const int chunk = idx & 7;        // 8 f16 per chunk
    const int t  = qb - 64 + row;
    const int tc = t < 0 ? 0 : (t >= SEQ ? SEQ - 1 : t);
    v8h v = ldg8(Kb, (unsigned)tc * HD + chunk * 8);
    *reinterpret_cast<v8h*>(&Ksh[row * KPAD + chunk * 8]) = v;
  }
  __syncthreads();

  // ---- Q A-fragments (row = r, d split into two K-steps of 32) ----
  const unsigned qo = (unsigned)(qw + r) * HD + half * 16;
  v16h aQ0 = cat16(ldg8(Qb, qo),      ldg8(Qb, qo + 8));
  v16h aQ1 = cat16(ldg8(Qb, qo + 32), ldg8(Qb, qo + 40));

  int rowok[8];
  #pragma unroll
  for (int rr = 0; rr < 8; rr++) rowok[rr] = (pm[qw + half * 8 + rr] != 0);

  // ---- scores: 10 tiles of 16x16, each = 2 WMMAs over d; B from LDS ----
  float sc[NT][8];
  #pragma unroll
  for (int kt = 0; kt < NT; kt++) {
    const int t = tstart + kt * 16 + r;  // this lane's key column
    const _Float16* kr = Ksh + (unsigned)(wid * 16 + kt * 16 + r) * KPAD + half * 16;
    v16h b0 = cat16(lds8(kr),      lds8(kr + 8));
    v16h b1 = cat16(lds8(kr + 32), lds8(kr + 40));
    v8f acc = {};
    acc = wmma16(aQ0, b0, acc);
    acc = wmma16(aQ1, b1, acc);
    const int tc  = t < 0 ? 0 : (t >= SEQ ? SEQ - 1 : t);
    const int tin = (int)((unsigned)t < (unsigned)SEQ) & (int)(pm[tc] != 0);
    #pragma unroll
    for (int rr = 0; rr < 8; rr++) {
      // |t - s| <= 64  <=>  0 <= kt*16 + cdiag - rr <= 128  (qw cancels)
      const int band = (int)((unsigned)(kt * 16 + cdiag - rr) <= 128u);
      const int v = tin & band & rowok[rr];
      sc[kt][rr] = v ? acc[rr] : NEG_INF;
    }
  }

  // ---- softmax per row: reduce across the 16 lanes of each half ----
  float inv_[8];
  #pragma unroll
  for (int rr = 0; rr < 8; rr++) {
    float m = NEG_INF;
    #pragma unroll
    for (int kt = 0; kt < NT; kt++) m = fmaxf(m, sc[kt][rr]);
    #pragma unroll
    for (int off = 1; off < 16; off <<= 1) m = fmaxf(m, __shfl_xor(m, off, 32));
    const float mm = (m > -1e30f) ? m : 0.f;
    float ssum = 0.f;
    #pragma unroll
    for (int kt = 0; kt < NT; kt++) {
      const float e = (sc[kt][rr] > -1e30f) ? __expf(sc[kt][rr] - mm) : 0.f;
      sc[kt][rr] = e;
      ssum += e;
    }
    #pragma unroll
    for (int off = 1; off < 16; off <<= 1) ssum += __shfl_xor(ssum, off, 32);
    inv_[rr] = (ssum > 0.f) ? 1.f / ssum : 0.f;  // all-masked row -> zeros
  }

  // all waves done reading the K tile -> safe to reuse the buffer for P
  __syncthreads();

  // ---- stage normalized P (f16) into per-wave LDS region ----
  _Float16* Pw = Pb + wid * (16 * PPAD);
  #pragma unroll
  for (int kt = 0; kt < NT; kt++) {
    #pragma unroll
    for (int rr = 0; rr < 8; rr++)
      Pw[(half * 8 + rr) * PPAD + kt * 16 + r] = (_Float16)(sc[kt][rr] * inv_[rr]);
  }
  // P region is wave-private; same-wave LDS ops are in-order -> no barrier.

  // ---- PV: 5 K-steps of 32 keys x 4 d-tiles of 16 = 20 WMMAs ----
  v8f accV[4] = {};
  #pragma unroll
  for (int j = 0; j < 5; j++) {
    const _Float16* pp = Pw + r * PPAD + j * 32 + half * 16;  // A-frag from LDS
    v16h aP = cat16(lds8(pp), lds8(pp + 8));
    const int t0  = tstart + j * 32 + half * 16;
    const int t0c = t0 < 0 ? 0 : (t0 > SEQ - 16 ? SEQ - 16 : t0);  // P==0 there
    #pragma unroll
    for (int dt = 0; dt < 4; dt++) {
      const unsigned vo = (unsigned)(dt * 16 + r) * SEQ + (unsigned)t0c;
      v16h bV = cat16(ldg8(Vb, vo), ldg8(Vb, vo + 8));
      accV[dt] = wmma16(aP, bV, accV[dt]);
    }
  }

  // ---- write vals[s][h*64+d] as f16 for the output GEMM ----
  #pragma unroll
  for (int dt = 0; dt < 4; dt++) {
    #pragma unroll
    for (int rr = 0; rr < 8; rr++) {
      const unsigned s = (unsigned)(qw + half * 8 + rr);
      vals[s * EMB + h * HD + dt * 16 + r] = (_Float16)accV[dt][rr];
    }
  }
}

// ---------------------------------------------------------------------------
extern "C" void kernel_launch(void* const* d_in, const int* in_sizes, int n_in,
                              void* d_out, int out_size, void* d_ws, size_t ws_size,
                              hipStream_t stream) {
  const float* x    = (const float*)d_in[0];
  const int*   pm   = (const int*)  d_in[1];
  const float* wqkv = (const float*)d_in[2];
  const float* bqkv = (const float*)d_in[3];
  const float* wo   = (const float*)d_in[4];
  const float* bo   = (const float*)d_in[5];
  float* out = (float*)d_out;

  char* ws = (char*)d_ws;
  auto alloc = [&](size_t bytes) {
    char* p = ws;
    ws += (bytes + 255) & ~(size_t)255;
    return p;
  };
  _Float16* x16  = (_Float16*)alloc((size_t)SEQ * EMB * 2);
  _Float16* wq16 = (_Float16*)alloc((size_t)3 * EMB * EMB * 2);
  _Float16* wo16 = (_Float16*)alloc((size_t)EMB * EMB * 2);
  _Float16* Qh   = (_Float16*)alloc((size_t)HEADS * SEQ * HD * 2);
  _Float16* Kh   = (_Float16*)alloc((size_t)HEADS * SEQ * HD * 2);
  _Float16* Vt   = (_Float16*)alloc((size_t)HEADS * HD * SEQ * 2);
  _Float16* vals = (_Float16*)alloc((size_t)SEQ * EMB * 2);

  cvt_f32_f16<<<dim3(256), dim3(256), 0, stream>>>(x, x16, SEQ * EMB / 8);
  cvt_f32_f16<<<dim3(256), dim3(256), 0, stream>>>(wqkv, wq16, 3 * EMB * EMB / 8);
  cvt_f32_f16<<<dim3(256), dim3(256), 0, stream>>>(wo, wo16, EMB * EMB / 8);

  // qkv = x @ wqkv^T + bqkv, scattered to Q (scaled), K, V^T
  gemm_wmma<1><<<dim3(SEQ / 128, (3 * EMB) / 64), dim3(256), 0, stream>>>(
      x16, EMB, wq16, EMB, bqkv, EMB, nullptr, 0, Qh, Kh, Vt, SEQ);

  attn_wmma<<<dim3(SEQ / 128, HEADS), dim3(256), 0, stream>>>(Qh, Kh, Vt, pm, vals);

  // out = vals @ wo^T + bo  (f32 result)
  gemm_wmma<0><<<dim3(SEQ / 128, EMB / 64), dim3(256), 0, stream>>>(
      vals, EMB, wo16, EMB, bo, EMB, out, EMB, nullptr, nullptr, nullptr, SEQ);
}
